// CFEM_19877108646076
// MI455X (gfx1250) — matrix operand order repaired and checked by reading, not compiled, not run on hardware
//
#include <hip/hip_runtime.h>
#include <hip/hip_bf16.h>

// ---------------------------------------------------------------------------
// CFEM on MI455X (gfx1250): all GEMM-shaped work via V_WMMA_F32_16X16X32_F16.
// f16 NHWC storage + fp32 accumulation. Wide-M waves (64 px x 16 ch, 4 accs
// per B fragment). Inputs pre-padded (halo=7) so every A load is
// unconditional and immediate-offset addressed from one base per row.
// ---------------------------------------------------------------------------

typedef __attribute__((ext_vector_type(16))) _Float16 v16h;
typedef __attribute__((ext_vector_type(8)))  _Float16 v8h;
typedef __attribute__((ext_vector_type(8)))  float    v8f;

union AFrag { v16h v; v8h h[2]; };

static __device__ __forceinline__ v8f wmma_f16(v16h a, v16h b, v8f c) {
    return __builtin_amdgcn_wmma_f32_16x16x32_f16(
        false, a, false, b, (short)0, c, false, false);
}

#define BB   4
#define HH   128
#define WW   128
#define CIN  128
#define CO   64
#define HP   130            // feats pad = 1
#define WP   130
#define PAD  7              // input pad = max dilation
#define HX   (HH + 2 * PAD) // 142
#define WX   (WW + 2 * PAD) // 142

// ---------------------------------------------------------------------------
// x: fp32 NCHW -> f16 NHWC into interior of zero-padded xpad (B,142,142,128)
// ---------------------------------------------------------------------------
__global__ void k_cast_x(const float* __restrict__ x, _Float16* __restrict__ xpad) {
    int i = blockIdx.x * blockDim.x + threadIdx.x;     // NCHW flat
    const int total = BB * CIN * HH * WW;
    if (i >= total) return;
    int xx = i & 127; int t = i >> 7;
    int y  = t & 127; t >>= 7;
    int c  = t & 127; int b = t >> 7;
    xpad[((size_t)((b * HX + y + PAD) * WX + xx + PAD)) * CIN + c] = (_Float16)x[i];
}

__global__ void k_zero8(_Float16* __restrict__ p, int n8) {
    int i = blockIdx.x * blockDim.x + threadIdx.x;
    if (i >= n8) return;
    v8h z = {};
    *(v8h*)(p + (size_t)i * 8) = z;
}

// ---------------------------------------------------------------------------
// Generic weight pack: OIHW fp32 -> B-fragment layout f16.
//   dst[((ntile*ksteps + s)*32 + lane)*16 + e]
//   n = ntile*16 + (lane&15); hi = lane>>4; klocal = (e&7) + hi*8 + (e>>3)*16;
//   s = tap*(I/32) + cb; c = cb*32 + klocal.  n >= O -> 0.
// ---------------------------------------------------------------------------
__global__ void k_pack(const float* __restrict__ w, _Float16* __restrict__ dst,
                       int O, int I, int taps, int total) {
    int i = blockIdx.x * blockDim.x + threadIdx.x;
    if (i >= total) return;
    int e    = i & 15;
    int lane = (i >> 4) & 31;
    int rest = i >> 9;
    int cbs  = I >> 5;
    int ksteps = taps * cbs;
    int s     = rest % ksteps;
    int n     = (rest / ksteps) * 16 + (lane & 15);
    int klocal = (e & 7) + (lane >> 4) * 8 + (e >> 3) * 16;
    int tap = s / cbs, cb = s % cbs;
    int c   = cb * 32 + klocal;
    float v = (n < O) ? w[((size_t)n * I + c) * taps + tap] : 0.0f;
    dst[i] = (_Float16)v;
}

// ---------------------------------------------------------------------------
// Branch conv: 3x3 dil=DIL (compile-time), 128 -> 64ch, +bias, *scale.
// Reads padded xpad: no bounds checks; all load offsets are immediates from
// one base per ky. Wave = 64 px x 16 ch; waves = 1024 mgroups * 4 ntiles.
// ---------------------------------------------------------------------------
template <int DIL>
__global__ void k_branch_conv(const _Float16* __restrict__ xpad,
                              const _Float16* __restrict__ wp,
                              const float* __restrict__ bias,
                              _Float16* __restrict__ fpad,
                              float scale) {
    const int lane  = threadIdx.x & 31;
    const int wave  = blockIdx.x * (blockDim.x >> 5) + (threadIdx.x >> 5);
    const int ntile = wave & 3;
    const int mg    = wave >> 2;
    const int r = lane & 15, hi = lane >> 4;
    const int pixb = mg * 64;                 // 64 pixels in one image row
    const int x0 = pixb & 127, y = (pixb >> 7) & 127, b = pixb >> 14;

    const _Float16* wptr = wp + (size_t)(ntile * 36) * 512 + lane * 16;
    v8f acc[4] = {{}, {}, {}, {}};
#pragma unroll
    for (int ky = 0; ky < 3; ++ky) {
        const _Float16* rowp =
            xpad + (size_t)((b * HX + (y + PAD + (ky - 1) * DIL)) * WX +
                            x0 + r + PAD) * CIN + hi * 8;
#pragma unroll
        for (int kx = 0; kx < 3; ++kx) {
            const int dxo = (kx - 1) * DIL * CIN;
            __builtin_prefetch(wptr + 4 * 512, 0, 1);
#pragma unroll
            for (int cb = 0; cb < 4; ++cb) {
                v16h bf = *(const v16h*)(wptr); wptr += 512;
#pragma unroll
                for (int j = 0; j < 4; ++j) {
                    const _Float16* src = rowp + (j * 16 * CIN + dxo + cb * 32);
                    AFrag a;
                    a.h[0] = *(const v8h*)(src);
                    a.h[1] = *(const v8h*)(src + 16);
                    acc[j] = wmma_f16(a.v, bf, acc[j]);
                }
            }
        }
    }
    float bn = bias[ntile * 16 + r];
#pragma unroll
    for (int j = 0; j < 4; ++j)
#pragma unroll
        for (int v = 0; v < 8; ++v) {
            int x2 = x0 + j * 16 + hi * 8 + v;
            size_t o = ((size_t)((b * HP + y + 1) * WP + x2 + 1)) * CO +
                       ntile * 16 + r;
            fpad[o] = (_Float16)((acc[j][v] + bn) * scale);
        }
}

// ---------------------------------------------------------------------------
// Offset conv: 3x3 pad=1, 64 -> 18 (N padded to 32). Reads padded feats: no
// bounds checks, immediate offsets. Wave = 64 px x 16 ch; waves = 1024 * 2.
// ---------------------------------------------------------------------------
__global__ void k_offset_conv(const _Float16* __restrict__ fpad,
                              const _Float16* __restrict__ wp,
                              const float* __restrict__ pb,
                              float* __restrict__ offs) {
    const int lane  = threadIdx.x & 31;
    const int wave  = blockIdx.x * (blockDim.x >> 5) + (threadIdx.x >> 5);
    const int ntile = wave & 1;
    const int mg    = wave >> 1;
    const int r = lane & 15, hi = lane >> 4;
    const int pixb = mg * 64;
    const int x0 = pixb & 127, y = (pixb >> 7) & 127, b = pixb >> 14;

    const _Float16* wptr = wp + (size_t)(ntile * 18) * 512 + lane * 16;
    v8f acc[4] = {{}, {}, {}, {}};
#pragma unroll
    for (int ky = 0; ky < 3; ++ky) {
        const _Float16* rowp =
            fpad + (size_t)((b * HP + y + ky) * WP + x0 + r) * CO + hi * 8;
#pragma unroll
        for (int kx = 0; kx < 3; ++kx) {
            __builtin_prefetch(wptr + 2 * 512, 0, 1);
#pragma unroll
            for (int cb = 0; cb < 2; ++cb) {
                v16h bf = *(const v16h*)(wptr); wptr += 512;
#pragma unroll
                for (int j = 0; j < 4; ++j) {
                    const _Float16* src = rowp + (j * 16 * CO + kx * CO + cb * 32);
                    AFrag a;
                    a.h[0] = *(const v8h*)(src);
                    a.h[1] = *(const v8h*)(src + 16);
                    acc[j] = wmma_f16(a.v, bf, acc[j]);
                }
            }
        }
    }
    int n = ntile * 16 + r;
    if (n < 18) {
        float bn = pb[n];
#pragma unroll
        for (int j = 0; j < 4; ++j)
#pragma unroll
            for (int v = 0; v < 8; ++v) {
                int p2 = pixb + j * 16 + hi * 8 + v;
                offs[(size_t)p2 * 18 + n] = acc[j][v] + bn;
            }
    }
}

// ---------------------------------------------------------------------------
// Deformable einsum: bilinear-sample padded feats at offset taps, WMMA vs dw.
// Coordinate math + 4 neighbor bases hoisted per tap; inner cb loop does the
// channel-vector loads, fp32 blend, WMMA. waves = 4096 mtiles * 4 ntiles.
// ---------------------------------------------------------------------------
__global__ void k_deform(const _Float16* __restrict__ fpad,
                         const float* __restrict__ offs,
                         const _Float16* __restrict__ wp,
                         const float* __restrict__ db,
                         _Float16* __restrict__ comb,
                         int chan_off) {
    const int lane  = threadIdx.x & 31;
    const int wave  = blockIdx.x * (blockDim.x >> 5) + (threadIdx.x >> 5);
    const int ntile = wave & 3;
    const int mtile = wave >> 2;
    const int row = lane & 15, hi = lane >> 4;
    const int pix = mtile * 16 + row;
    const int xx = pix & 127, y = (pix >> 7) & 127, b = pix >> 14;
    const _Float16* base = fpad + (size_t)b * HP * WP * CO;

    const _Float16* wptr = wp + (size_t)(ntile * 18) * 512 + lane * 16;
    v8f acc = {};
#pragma unroll
    for (int ky = 0; ky < 3; ++ky) {
#pragma unroll
        for (int kx = 0; kx < 3; ++kx) {
            const int tap = ky * 3 + kx;
            float offy = offs[(size_t)pix * 18 + tap];
            float offx = offs[(size_t)pix * 18 + 9 + tap];
            float py = (float)(y + ky) + offy;     // (y+1)+(ky-1)
            float px = (float)(xx + kx) + offx;
            float fy = floorf(py), fx = floorf(px);
            float lty = fminf(fmaxf(fy, 0.f), 129.f);
            float ltx = fminf(fmaxf(fx, 0.f), 129.f);
            float rby = fminf(fmaxf(fy + 1.f, 0.f), 129.f);
            float rbx = fminf(fmaxf(fx + 1.f, 0.f), 129.f);
            float pyc = fminf(fmaxf(py, 0.f), 129.f);
            float pxc = fminf(fmaxf(px, 0.f), 129.f);
            float glt = (1.f + (lty - pyc)) * (1.f + (ltx - pxc));
            float grb = (1.f - (rby - pyc)) * (1.f - (rbx - pxc));
            float glb = (1.f + (lty - pyc)) * (1.f - (rbx - pxc));
            float grt = (1.f - (rby - pyc)) * (1.f + (ltx - pxc));
            int iy0 = (int)lty, ix0 = (int)ltx, iy1 = (int)rby, ix1 = (int)rbx;
            const _Float16* p00 = base + ((size_t)(iy0 * WP + ix0)) * CO + hi * 8;
            const _Float16* p01 = base + ((size_t)(iy0 * WP + ix1)) * CO + hi * 8;
            const _Float16* p10 = base + ((size_t)(iy1 * WP + ix0)) * CO + hi * 8;
            const _Float16* p11 = base + ((size_t)(iy1 * WP + ix1)) * CO + hi * 8;
#pragma unroll
            for (int cb = 0; cb < 2; ++cb) {
                v16h bf = *(const v16h*)(wptr); wptr += 512;
                AFrag a;
#pragma unroll
                for (int rr = 0; rr < 2; ++rr) {
                    const int c0 = cb * 32 + rr * 16;
                    v8h s00 = *(const v8h*)(p00 + c0);
                    v8h s01 = *(const v8h*)(p01 + c0);
                    v8h s10 = *(const v8h*)(p10 + c0);
                    v8h s11 = *(const v8h*)(p11 + c0);
#pragma unroll
                    for (int e = 0; e < 8; ++e) {
                        float sv = glt * (float)s00[e] + grb * (float)s11[e] +
                                   glb * (float)s01[e] + grt * (float)s10[e];
                        a.v[rr * 8 + e] = (_Float16)sv;
                    }
                }
                acc = wmma_f16(a.v, bf, acc);
            }
        }
    }
    int n = ntile * 16 + row;
    float bn = db[n];
#pragma unroll
    for (int v = 0; v < 8; ++v) {
        int p2 = mtile * 16 + hi * 8 + v;
        comb[(size_t)p2 * 256 + chan_off + n] = (_Float16)(acc[v] + bn);
    }
}

// ---------------------------------------------------------------------------
// Final: concat(256) x cw(64x256) 1x1 + bias + ReLU -> fp32 NCHW output.
// Per-j base pointers hoisted; s*32 offsets are immediates. waves = 1024 * 4.
// ---------------------------------------------------------------------------
__global__ void k_final(const _Float16* __restrict__ comb,
                        const _Float16* __restrict__ wp,
                        const float* __restrict__ cb,
                        float* __restrict__ out) {
    const int lane  = threadIdx.x & 31;
    const int wave  = blockIdx.x * (blockDim.x >> 5) + (threadIdx.x >> 5);
    const int ntile = wave & 3;
    const int mg    = wave >> 2;
    const int r = lane & 15, hi = lane >> 4;
    const int pixb = mg * 64;
    const int x0 = pixb & 127, y = (pixb >> 7) & 127, b = pixb >> 14;

    const _Float16* wptr = wp + (size_t)(ntile * 8) * 512 + lane * 16;
    const _Float16* srcj[4];
#pragma unroll
    for (int j = 0; j < 4; ++j)
        srcj[j] = comb + (size_t)(pixb + j * 16 + r) * 256 + hi * 8;

    v8f acc[4] = {{}, {}, {}, {}};
#pragma unroll
    for (int s = 0; s < 8; ++s) {
        v16h bf = *(const v16h*)(wptr); wptr += 512;
#pragma unroll
        for (int j = 0; j < 4; ++j) {
            AFrag a;
            a.h[0] = *(const v8h*)(srcj[j] + s * 32);
            a.h[1] = *(const v8h*)(srcj[j] + s * 32 + 16);
            acc[j] = wmma_f16(a.v, bf, acc[j]);
        }
    }
    int n = ntile * 16 + r;
    float bn = cb[n];
#pragma unroll
    for (int j = 0; j < 4; ++j)
#pragma unroll
        for (int v = 0; v < 8; ++v) {
            int x2 = x0 + j * 16 + hi * 8 + v;
            float rv = fmaxf(acc[j][v] + bn, 0.0f);
            out[(((size_t)(b * CO + n) * HH + y) * WW) + x2] = rv;
        }
}

// ---------------------------------------------------------------------------
// Host: workspace layout + launch sequence (all on `stream`, graph-safe)
// ---------------------------------------------------------------------------
extern "C" void kernel_launch(void* const* d_in, const int* in_sizes, int n_in,
                              void* d_out, int out_size, void* d_ws, size_t ws_size,
                              hipStream_t stream) {
    (void)in_sizes; (void)n_in; (void)out_size; (void)ws_size;
    const float* x   = (const float*)d_in[0];
    const float* wB[4] = {(const float*)d_in[1], (const float*)d_in[3],
                          (const float*)d_in[5], (const float*)d_in[7]};
    const float* bVB[4] = {(const float*)d_in[2], (const float*)d_in[4],
                           (const float*)d_in[6], (const float*)d_in[8]};
    const float* p_w = (const float*)d_in[9];
    const float* p_b = (const float*)d_in[10];
    const float* dw  = (const float*)d_in[11];
    const float* db  = (const float*)d_in[12];
    const float* cw  = (const float*)d_in[13];
    const float* cb  = (const float*)d_in[14];

    const size_t XPELEMS = (size_t)BB * HX * WX * CIN;   // padded input halfs
    const size_t FPELEMS = (size_t)BB * HP * WP * CO;    // padded feats halfs
    const size_t OFELEMS = (size_t)BB * HH * WW * 18;    // offset floats
    char* ws = (char*)d_ws;
    size_t off = 0;
    _Float16* xpad = (_Float16*)(ws + off); off += XPELEMS * 2;
    _Float16* fpad = (_Float16*)(ws + off); off += FPELEMS * 4 * 2;
    float*    offs = (float*)   (ws + off); off += OFELEMS * 4 * 4;
    _Float16* comb = (_Float16*)(ws + off); off += (size_t)BB*HH*WW*256*2;
    _Float16* wpk  = (_Float16*)(ws + off); off += (size_t)4*4*36*512*2;
    _Float16* ppk  = (_Float16*)(ws + off); off += (size_t)2*18*512*2;
    _Float16* dpk  = (_Float16*)(ws + off); off += (size_t)4*18*512*2;
    _Float16* cpk  = (_Float16*)(ws + off); off += (size_t)4*8*512*2;

    for (int br = 0; br < 4; ++br)
        hipLaunchKernelGGL(k_pack, dim3(288), dim3(256), 0, stream,
                           wB[br], wpk + (size_t)br * 73728, 64, 128, 9, 73728);
    hipLaunchKernelGGL(k_pack, dim3(72),  dim3(256), 0, stream, p_w, ppk, 18, 64, 9, 18432);
    hipLaunchKernelGGL(k_pack, dim3(144), dim3(256), 0, stream, dw,  dpk, 64, 64, 9, 36864);
    hipLaunchKernelGGL(k_pack, dim3(64),  dim3(256), 0, stream, cw,  cpk, 64, 256, 1, 16384);

    // zero xpad + fpad (adjacent region), then fill interiors
    {
        const int n8 = (int)((XPELEMS + FPELEMS * 4) / 8);
        hipLaunchKernelGGL(k_zero8, dim3((n8 + 255) / 256), dim3(256), 0, stream,
                           xpad, n8);
    }
    hipLaunchKernelGGL(k_cast_x, dim3(32768), dim3(256), 0, stream, x, xpad);

    const float scl[4] = {1.f, 3.f, 5.f, 7.f};
    hipLaunchKernelGGL(k_branch_conv<1>, dim3(512), dim3(256), 0, stream,
                       xpad, wpk + (size_t)0 * 73728, bVB[0],
                       fpad + (size_t)0 * FPELEMS, scl[0]);
    hipLaunchKernelGGL(k_branch_conv<3>, dim3(512), dim3(256), 0, stream,
                       xpad, wpk + (size_t)1 * 73728, bVB[1],
                       fpad + (size_t)1 * FPELEMS, scl[1]);
    hipLaunchKernelGGL(k_branch_conv<5>, dim3(512), dim3(256), 0, stream,
                       xpad, wpk + (size_t)2 * 73728, bVB[2],
                       fpad + (size_t)2 * FPELEMS, scl[2]);
    hipLaunchKernelGGL(k_branch_conv<7>, dim3(512), dim3(256), 0, stream,
                       xpad, wpk + (size_t)3 * 73728, bVB[3],
                       fpad + (size_t)3 * FPELEMS, scl[3]);

    for (int br = 0; br < 4; ++br)
        hipLaunchKernelGGL(k_offset_conv, dim3(256), dim3(256), 0, stream,
                           fpad + (size_t)br * FPELEMS, ppk, p_b,
                           offs + (size_t)br * OFELEMS);

    for (int br = 0; br < 4; ++br)
        hipLaunchKernelGGL(k_deform, dim3(2048), dim3(256), 0, stream,
                           fpad + (size_t)br * FPELEMS, offs + (size_t)br * OFELEMS,
                           dpk, db, comb, br * 64);

    hipLaunchKernelGGL(k_final, dim3(512), dim3(256), 0, stream,
                       comb, cpk, cb, (float*)d_out);
}
